// RecurrentBackbone_11656541241806
// MI455X (gfx1250) — matrix operand order repaired and checked by reading, not compiled, not run on hardware
//
#include <hip/hip_runtime.h>
#include <math.h>

// Problem dimensions (match reference).
#define B_  256
#define T_  256
#define D_  512
#define FC_ 512
#define H_  512
#define G3_ (3 * H_)
#define K_  512   // all GEMMs reduce over 512
static constexpr float LN_EPS = 1e-5f;

typedef __attribute__((ext_vector_type(16))) __bf16 v16bf;
typedef __attribute__((ext_vector_type(8)))  float  v8f;
typedef __attribute__((ext_vector_type(4)))  float  f32x4;

union FragU { v16bf v; f32x4 f[2]; };

// Scheduler pipelining hints (no-op if builtin unavailable).
#if defined(__has_builtin)
#if __has_builtin(__builtin_amdgcn_sched_group_barrier)
#define SCHED_GROUP(mask, cnt) __builtin_amdgcn_sched_group_barrier((mask), (cnt), 0)
#endif
#endif
#ifndef SCHED_GROUP
#define SCHED_GROUP(mask, cnt)
#endif
// mask bits: 0x008 = MFMA/WMMA, 0x020 = VMEM read, 0x100 = DS read.

// ---------------------------------------------------------------------------
// LDS A-panel geometry: 32 rows x 512 halves, rows padded by 8 halves (16B)
// so row stride = 1040B = 260 dwords == 4 (mod 64 banks): lane r of a
// fragment read starts at bank 4r -> the 16 lanes cover all 64 banks with
// no conflicts, and 16B alignment for ds_load_b128 is preserved.
// ---------------------------------------------------------------------------
#define PANEL_ROWS 32
#define LDSP (K_ + 8)
#define PANEL_ELTS (PANEL_ROWS * LDSP)

// ---------------------------------------------------------------------------
// WMMA fragment loaders (CDNA5 ISA 7.12.2 layouts, wave32)
// ---------------------------------------------------------------------------

// 16-bit A matrix 16x32 (MxK) from the padded LDS tile (row stride LDSP).
// lane<16 : M=lane,     K = {k0..k0+7, k0+16..k0+23}
// lane>=16: M=lane-16,  K = {k0+8..k0+15, k0+24..k0+31}
__device__ inline v16bf load_frag_a_lds(const __bf16* tile, int lane) {
    int r  = lane & 15;
    int hi = lane >> 4;
    const __bf16* p = tile + r * LDSP + (hi ? 8 : 0);
    FragU u;
    u.f[0] = *(const f32x4*)(p);        // 8 halves
    u.f[1] = *(const f32x4*)(p + 16);   // 8 halves, 16 halves further in K
    return u.v;
}

// 16-bit B matrix 32x16 (KxN), supplied as W[N,K] row-major (B[k][n]=W[n][k]).
// tile -> &W[n0*ldk + k0].
// lane<16 : N=lane,     K = {k0..k0+7, k0+16..k0+23}
// lane>=16: N=lane-16,  K = {k0+8..k0+15, k0+24..k0+31}
__device__ inline v16bf load_frag_b(const __bf16* tile, int ldk, int lane) {
    int r  = lane & 15;
    int hi = lane >> 4;
    const __bf16* p = tile + (size_t)r * ldk + (hi ? 8 : 0);
    FragU u;
    u.f[0] = *(const f32x4*)(p);
    u.f[1] = *(const f32x4*)(p + 16);
    return u.v;
}

__device__ inline v8f wmma_bf16(v16bf a, v16bf b, v8f c) {
    return __builtin_amdgcn_wmma_f32_16x16x32_bf16(
        /*neg_a=*/false, a, /*neg_b=*/false, b,
        /*c_mod=*/(short)0, c, /*reuse_a=*/false, /*reuse_b=*/false);
}

__device__ inline float sigmoidf_fast(float x) {
    return 1.0f / (1.0f + __expf(-x));
}

// ---------------------------------------------------------------------------
// CDNA5 async global->LDS staging (ASYNCcnt path, inline asm).
// Each active lane moves 16B from its global address to its own LDS offset;
// completion tracked with ASYNCcnt -> s_wait_asynccnt, then a workgroup
// barrier makes the panel visible to all 8 waves.
// ---------------------------------------------------------------------------
__device__ inline void stage_a_panel(const __bf16* __restrict__ A, int m0,
                                     __bf16* smem) {
    const __bf16* src = A + (size_t)m0 * K_;
    unsigned lbase = (unsigned)(unsigned long long)(void*)smem;  // LDS aperture low bits
    // 32 rows x 64 b128-transfers per row = 2048 transfers, 8 per thread.
    for (int i = threadIdx.x; i < PANEL_ROWS * (K_ / 8); i += 256) {
        int row  = i >> 6;          // 64 x b128 per 512-half row
        int col8 = i & 63;          // b128 index within row
        unsigned loff = lbase + (unsigned)(row * LDSP + col8 * 8) * 2u;
        const __bf16* g = src + (size_t)row * K_ + col8 * 8;
        asm volatile("global_load_async_to_lds_b128 %0, %1, off"
                     :: "v"(loff), "v"(g)
                     : "memory");
    }
    asm volatile("s_wait_asynccnt 0x0" ::: "memory");
    __syncthreads();
}

// ---------------------------------------------------------------------------
// Shared GEMM core: per-wave 16x64 output (4 N-tiles), A from LDS panel,
// B (weights, L1/L2-hot) double-buffered in registers across K chunks.
// sched_group_barrier enforces issue order per iteration:
//   ds reads (A_i) -> global reads (B_{i+1}) -> 4x WMMA (consume B_i),
// so the s_wait before each WMMA keeps the next chunk's loads in flight.
// ---------------------------------------------------------------------------
__device__ inline void gemm_core(const __bf16* __restrict__ W,
                                 const __bf16* smem,
                                 int m_local, int nb, int lane, v8f acc[4]) {
    v16bf bcur[4];
#pragma unroll
    for (int j = 0; j < 4; ++j)
        bcur[j] = load_frag_b(W + (size_t)(nb + j * 16) * K_, K_, lane);
    for (int k0 = 0; k0 < K_; k0 += 32) {
        const bool more = (k0 + 32) < K_;
        v16bf a = load_frag_a_lds(smem + m_local * LDSP + k0, lane);
        v16bf bn[4];
        if (more) {
#pragma unroll
            for (int j = 0; j < 4; ++j)
                bn[j] = load_frag_b(W + (size_t)(nb + j * 16) * K_ + k0 + 32, K_, lane);
        }
#pragma unroll
        for (int j = 0; j < 4; ++j)
            acc[j] = wmma_bf16(a, bcur[j], acc[j]);
        SCHED_GROUP(0x100, 2);   // 2 ds_load_b128 (A chunk i)
        SCHED_GROUP(0x020, 8);   // 8 global_load_b128 (B chunk i+1)
        SCHED_GROUP(0x008, 4);   // 4 WMMAs (consume chunk i)
        if (more) {
#pragma unroll
            for (int j = 0; j < 4; ++j) bcur[j] = bn[j];
        }
    }
}

// ---------------------------------------------------------------------------
// Small helper kernels
// ---------------------------------------------------------------------------

__global__ void k_cvt_bf16(const float* __restrict__ in, __bf16* __restrict__ out, int n) {
    int i = blockIdx.x * blockDim.x + threadIdx.x;
    if (i < n) out[i] = (__bf16)in[i];
}

// LayerNorm over D=512 per row, output bf16. One 256-thread block per row.
__global__ void k_layernorm(const float* __restrict__ x,
                            const float* __restrict__ gamma,
                            const float* __restrict__ beta,
                            __bf16* __restrict__ out) {
    __shared__ float s1[256];
    __shared__ float s2[256];
    const int row = blockIdx.x;
    const int tid = threadIdx.x;
    const float* p = x + (size_t)row * D_;
    float a = p[tid];
    float b = p[tid + 256];
    s1[tid] = a + b;
    s2[tid] = a * a + b * b;
    __syncthreads();
    for (int s = 128; s > 0; s >>= 1) {
        if (tid < s) { s1[tid] += s1[tid + s]; s2[tid] += s2[tid + s]; }
        __syncthreads();
    }
    const float mean = s1[0] * (1.0f / D_);
    const float var  = s2[0] * (1.0f / D_) - mean * mean;
    const float inv  = rsqrtf(var + LN_EPS);
    __bf16* o = out + (size_t)row * D_;
    o[tid]       = (__bf16)((a - mean) * inv * gamma[tid] + beta[tid]);
    o[tid + 256] = (__bf16)((b - mean) * inv * gamma[tid + 256] + beta[tid + 256]);
}

// ---------------------------------------------------------------------------
// GEMM kernels: 256 threads = 8 waves; block tile 32 rows x 256 cols.
// Wave w: M-stripe (w>>2)*16, N-group (w&3)*64 (4 16x16 tiles).
// A panel async-staged to LDS once per block.
// ---------------------------------------------------------------------------

// x = relu(xn @ w_in^T + b_in), bf16 out.
__global__ void k_gemm_relu_bf16(const __bf16* __restrict__ A,
                                 const __bf16* __restrict__ W,
                                 const float* __restrict__ bias,
                                 __bf16* __restrict__ out,
                                 int N) {
    __shared__ __bf16 smem[PANEL_ELTS];
    const int lane = threadIdx.x & 31;
    const int wave = threadIdx.x >> 5;
    const int m0 = blockIdx.y * PANEL_ROWS;
    const int ml = (wave >> 2) * 16;
    const int nb = blockIdx.x * 256 + (wave & 3) * 64;
    stage_a_panel(A, m0, smem);
    v8f acc[4] = {};
    gemm_core(W, smem, ml, nb, lane, acc);
    const int hi = lane >> 4;
    const int nl = lane & 15;
#pragma unroll
    for (int j = 0; j < 4; ++j) {
        const int col = nb + j * 16 + nl;
        const float bv = bias[col];
#pragma unroll
        for (int r = 0; r < 8; ++r) {
            const int row = m0 + ml + r + hi * 8;
            float v = acc[j][r] + bv;
            v = v > 0.0f ? v : 0.0f;
            out[(size_t)row * N + col] = (__bf16)v;
        }
    }
}

// gi = x @ w_ih^T + b_ih, fp32 out (gate pre-activations in full precision).
__global__ void k_gemm_bias_f32(const __bf16* __restrict__ A,
                                const __bf16* __restrict__ W,
                                const float* __restrict__ bias,
                                float* __restrict__ out,
                                int N) {
    __shared__ __bf16 smem[PANEL_ELTS];
    const int lane = threadIdx.x & 31;
    const int wave = threadIdx.x >> 5;
    const int m0 = blockIdx.y * PANEL_ROWS;
    const int ml = (wave >> 2) * 16;
    const int nb = blockIdx.x * 256 + (wave & 3) * 64;
    stage_a_panel(A, m0, smem);
    v8f acc[4] = {};
    gemm_core(W, smem, ml, nb, lane, acc);
    const int hi = lane >> 4;
    const int nl = lane & 15;
#pragma unroll
    for (int j = 0; j < 4; ++j) {
        const int col = nb + j * 16 + nl;
        const float bv = bias[col];
#pragma unroll
        for (int r = 0; r < 8; ++r) {
            const int row = m0 + ml + r + hi * 8;
            out[(size_t)row * N + col] = acc[j][r] + bv;
        }
    }
}

// outputs[b,t,:] = relu(hs[t,b,:] @ w_out^T + b_out); A rows are [T,B] order,
// stores scattered to [B,T,H] in d_out.
__global__ void k_gemm_out(const __bf16* __restrict__ A,
                           const __bf16* __restrict__ W,
                           const float* __restrict__ bias,
                           float* __restrict__ out) {
    __shared__ __bf16 smem[PANEL_ELTS];
    const int lane = threadIdx.x & 31;
    const int wave = threadIdx.x >> 5;
    const int m0 = blockIdx.y * PANEL_ROWS;
    const int ml = (wave >> 2) * 16;
    const int nb = blockIdx.x * 256 + (wave & 3) * 64;
    stage_a_panel(A, m0, smem);
    v8f acc[4] = {};
    gemm_core(W, smem, ml, nb, lane, acc);
    const int hi = lane >> 4;
    const int nl = lane & 15;
#pragma unroll
    for (int j = 0; j < 4; ++j) {
        const int col = nb + j * 16 + nl;
        const float bv = bias[col];
#pragma unroll
        for (int r = 0; r < 8; ++r) {
            const int row = m0 + ml + r + hi * 8;   // row = t*B + b
            const int t = row / B_;
            const int b = row - t * B_;
            float v = acc[j][r] + bv;
            v = v > 0.0f ? v : 0.0f;
            out[((size_t)b * T_ + t) * H_ + col] = v;
        }
    }
}

// ---------------------------------------------------------------------------
// One GRU timestep, fused: gh = (h*mask) @ w_hh^T + b_hh for all 3 gates,
// gate nonlinearities against precomputed gi, write h_out (fp32) + hs (bf16).
// Block: 128 threads = 4 waves, each wave one 16x16 h-tile with 3 accumulators.
// Next K-chunk's h loads + 3 weight fragments are double-buffered in registers.
// Grid: (H/64, B/16).
// ---------------------------------------------------------------------------
__global__ void k_gru_step(const float* __restrict__ h_in,
                           float* __restrict__ h_out,
                           const __bf16* __restrict__ Whh,
                           const float* __restrict__ b_hh,
                           const float* __restrict__ gi,
                           const int* __restrict__ ep,
                           __bf16* __restrict__ hs,
                           int t) {
    const int lane = threadIdx.x & 31;
    const int wave = threadIdx.x >> 5;
    const int n0 = blockIdx.x * 64 + wave * 16;
    const int m0 = blockIdx.y * 16;
    const int rl = lane & 15;
    const int hi = lane >> 4;

    // A-fragment source: masked h row (built on the fly fp32 -> bf16).
    const int arow = m0 + rl;
    const float mval = 1.0f - (float)ep[arow * T_ + t];
    const float* hp = h_in + (size_t)arow * H_;
    const int kboff = hi ? 8 : 0;

    v8f accR = {}, accZ = {}, accN = {};

    // Prologue loads (chunk 0).
    f32x4 c0 = *(const f32x4*)(hp + kboff);
    f32x4 c1 = *(const f32x4*)(hp + kboff + 4);
    f32x4 c2 = *(const f32x4*)(hp + kboff + 16);
    f32x4 c3 = *(const f32x4*)(hp + kboff + 20);
    v16bf bR = load_frag_b(Whh + (size_t)(n0) * H_, H_, lane);
    v16bf bZ = load_frag_b(Whh + (size_t)(n0 + H_) * H_, H_, lane);
    v16bf bN = load_frag_b(Whh + (size_t)(n0 + 2 * H_) * H_, H_, lane);

    for (int k0 = 0; k0 < H_; k0 += 32) {
        const bool more = (k0 + 32) < H_;
        FragU ua;
#pragma unroll
        for (int i = 0; i < 4; ++i) {
            ua.v[i]      = (__bf16)(c0[i] * mval);
            ua.v[4 + i]  = (__bf16)(c1[i] * mval);
            ua.v[8 + i]  = (__bf16)(c2[i] * mval);
            ua.v[12 + i] = (__bf16)(c3[i] * mval);
        }
        f32x4 n0f, n1f, n2f, n3f;
        v16bf nR, nZ, nN;
        if (more) {
            const int kb = k0 + 32 + kboff;
            n0f = *(const f32x4*)(hp + kb);
            n1f = *(const f32x4*)(hp + kb + 4);
            n2f = *(const f32x4*)(hp + kb + 16);
            n3f = *(const f32x4*)(hp + kb + 20);
            nR = load_frag_b(Whh + (size_t)(n0) * H_ + k0 + 32, H_, lane);
            nZ = load_frag_b(Whh + (size_t)(n0 + H_) * H_ + k0 + 32, H_, lane);
            nN = load_frag_b(Whh + (size_t)(n0 + 2 * H_) * H_ + k0 + 32, H_, lane);
        }
        accR = wmma_bf16(ua.v, bR, accR);
        accZ = wmma_bf16(ua.v, bZ, accZ);
        accN = wmma_bf16(ua.v, bN, accN);
        SCHED_GROUP(0x020, 10);  // next chunk's h (4) + weight (6) loads first
        SCHED_GROUP(0x008, 3);   // then the 3 WMMAs of this chunk
        if (more) {
            c0 = n0f; c1 = n1f; c2 = n2f; c3 = n3f;
            bR = nR; bZ = nZ; bN = nN;
        }
    }

    const int col = n0 + rl;
    const float bRv = b_hh[col];
    const float bZv = b_hh[col + H_];
    const float bNv = b_hh[col + 2 * H_];
#pragma unroll
    for (int r = 0; r < 8; ++r) {
        const int row = m0 + r + hi * 8;
        const float* gr = gi + ((size_t)row * T_ + t) * G3_;
        const float ir  = gr[col];
        const float iz  = gr[col + H_];
        const float inn = gr[col + 2 * H_];
        const float rg = sigmoidf_fast(ir + accR[r] + bRv);
        const float zg = sigmoidf_fast(iz + accZ[r] + bZv);
        const float ng = tanhf(inn + rg * (accN[r] + bNv));
        const float mr = 1.0f - (float)ep[row * T_ + t];
        const float hm = h_in[(size_t)row * H_ + col] * mr;
        const float hv = (1.0f - zg) * ng + zg * hm;
        h_out[(size_t)row * H_ + col] = hv;
        hs[((size_t)t * B_ + row) * H_ + col] = (__bf16)hv;
    }
}

// ---------------------------------------------------------------------------
// Launch
// ---------------------------------------------------------------------------
extern "C" void kernel_launch(void* const* d_in, const int* in_sizes, int n_in,
                              void* d_out, int out_size, void* d_ws, size_t ws_size,
                              hipStream_t stream) {
    (void)in_sizes; (void)n_in; (void)out_size; (void)ws_size;

    const float* inputs    = (const float*)d_in[0];   // [B,T,D]
    const float* hidden0   = (const float*)d_in[1];   // [B,H]
    const int*   ep        = (const int*)  d_in[2];   // [B,T]
    const float* ln_gamma  = (const float*)d_in[3];
    const float* ln_beta   = (const float*)d_in[4];
    const float* w_in      = (const float*)d_in[5];   // [FC,D]
    const float* b_in      = (const float*)d_in[6];
    const float* w_ih      = (const float*)d_in[7];   // [3H,FC]
    const float* b_ih      = (const float*)d_in[8];
    const float* w_hh      = (const float*)d_in[9];   // [3H,H]
    const float* b_hh      = (const float*)d_in[10];
    const float* w_out     = (const float*)d_in[11];  // [H,H]
    const float* b_out     = (const float*)d_in[12];
    float* out = (float*)d_out;                       // [B,T,H] + [B,H]

    const size_t BT = (size_t)B_ * T_;
    char* ws = (char*)d_ws;
    size_t off = 0;
    auto alloc = [&](size_t bytes) { char* p = ws + off; off += (bytes + 255) & ~(size_t)255; return p; };

    __bf16* xn_bf    = (__bf16*)alloc(BT * D_ * 2);
    __bf16* w_in_bf  = (__bf16*)alloc((size_t)FC_ * D_ * 2);
    __bf16* w_ih_bf  = (__bf16*)alloc((size_t)G3_ * FC_ * 2);
    __bf16* w_hh_bf  = (__bf16*)alloc((size_t)G3_ * H_ * 2);
    __bf16* w_out_bf = (__bf16*)alloc((size_t)H_ * H_ * 2);
    __bf16* x_bf     = (__bf16*)alloc(BT * FC_ * 2);
    __bf16* hs_bf    = (__bf16*)alloc(BT * H_ * 2);
    float*  h_ping   = (float*)alloc((size_t)B_ * H_ * 4);
    float*  h_pong   = (float*)alloc((size_t)B_ * H_ * 4);
    float*  gi       = (float*)alloc(BT * (size_t)G3_ * 4);

    // 1) Weight conversions fp32 -> bf16.
    auto cvt = [&](const float* src, __bf16* dst, int n) {
        k_cvt_bf16<<<(n + 255) / 256, 256, 0, stream>>>(src, dst, n);
    };
    cvt(w_in,  w_in_bf,  FC_ * D_);
    cvt(w_ih,  w_ih_bf,  G3_ * FC_);
    cvt(w_hh,  w_hh_bf,  G3_ * H_);
    cvt(w_out, w_out_bf, H_ * H_);

    // 2) LayerNorm -> bf16.
    k_layernorm<<<(int)BT, 256, 0, stream>>>(inputs, ln_gamma, ln_beta, xn_bf);

    // 3) x = relu(xn @ w_in^T + b_in)  [BT, FC] bf16.
    k_gemm_relu_bf16<<<dim3(FC_ / 256, (int)(BT / PANEL_ROWS)), 256, 0, stream>>>(
        xn_bf, w_in_bf, b_in, x_bf, FC_);

    // 4) gi = x @ w_ih^T + b_ih  [BT, 3H] fp32.
    k_gemm_bias_f32<<<dim3(G3_ / 256, (int)(BT / PANEL_ROWS)), 256, 0, stream>>>(
        x_bf, w_ih_bf, b_ih, gi, G3_);

    // 5) h0 = hidden_state (do not mutate inputs).
    hipMemcpyAsync(h_ping, hidden0, (size_t)B_ * H_ * 4, hipMemcpyDeviceToDevice, stream);

    // 6) Sequential GRU scan: 256 step kernels, ping-pong h.
    float* hin = h_ping;
    float* hout = h_pong;
    for (int t = 0; t < T_; ++t) {
        k_gru_step<<<dim3(H_ / 64, B_ / 16), 128, 0, stream>>>(
            hin, hout, w_hh_bf, b_hh, gi, ep, hs_bf, t);
        float* tmp = hin; hin = hout; hout = tmp;
    }
    // After an even number of steps, final h lives in h_ping (== hin now).

    // 7) outputs = relu(hs @ w_out^T + b_out), scattered [T,B]->[B,T].
    k_gemm_out<<<dim3(H_ / 256, (int)(BT / PANEL_ROWS)), 256, 0, stream>>>(
        hs_bf, w_out_bf, b_out, out);

    // 8) h_last -> tail of d_out.
    hipMemcpyAsync(out + BT * H_, hin, (size_t)B_ * H_ * 4,
                   hipMemcpyDeviceToDevice, stream);
}
